// RetBlock_3685081940154
// MI455X (gfx1250) — matrix operand order repaired and checked by reading, not compiled, not run on hardware
//
#include <hip/hip_runtime.h>

// ---------------------------------------------------------------------------
// MI455X (gfx1250) RetBlock: bf16 WMMA for all matmuls, f32 accumulate.
// GEMM: async global->LDS staging (ASYNCcnt) + double-buffered LDS; all
// fragments are contiguous ds_load_b128 pairs. Attention: V staged transposed
// in LDS, O written back via vectorized b128 stores.
// ---------------------------------------------------------------------------

typedef __attribute__((ext_vector_type(16))) __bf16        v16bf;
typedef __attribute__((ext_vector_type(8)))  float         v8f;
typedef __attribute__((ext_vector_type(8)))  unsigned int  v8u;

static __device__ __forceinline__ unsigned short f2bf_raw(float f) {
  unsigned u = __builtin_bit_cast(unsigned, f);
  unsigned r = u + 0x7FFFu + ((u >> 16) & 1u);   // round-to-nearest-even
  return (unsigned short)(r >> 16);
}
static __device__ __forceinline__ float bf2f(unsigned short h) {
  return __builtin_bit_cast(float, ((unsigned)h) << 16);
}

static __device__ __forceinline__ v8f wmma_bf16(v16bf a, v16bf b, v8f c) {
  return __builtin_amdgcn_wmma_f32_16x16x32_bf16(
      /*neg_a=*/false, a, /*neg_b=*/false, b,
      /*c_mod=*/(short)0, c, /*reuse_a=*/false, /*reuse_b=*/false);
}

// Async 16B copy global -> LDS (GVS mode: SGPR64 base + VGPR byte offset).
// Tracked by ASYNCcnt; completion enforced with s_wait_asynccnt.
static __device__ __forceinline__ void async_ld_b128(unsigned lds_off,
                                                     unsigned long long sbase,
                                                     unsigned voff) {
  asm volatile("global_load_async_to_lds_b128 %0, %1, %2"
               :: "v"(lds_off), "v"(voff), "s"(sbase)
               : "memory");
}
static __device__ __forceinline__ void wait_async0() {
  asm volatile("s_wait_asynccnt 0" ::: "memory");
}

// A fragment (16x32 bf16) per ISA 7.12.2: lane<16 holds row M=lane, K in
// {0..7,16..23}; lane>=16 holds same rows, K in {8..15,24..31}.
// Source row-major, leading dim `ld` elements. Two contiguous 16B loads.
static __device__ __forceinline__ v16bf load_a_frag(
    const unsigned short* A, int ld, int row0, int k0, int lane) {
  int half = (lane >> 4) & 1;
  const unsigned short* p = A + (size_t)(row0 + (lane & 15)) * ld + k0 + half * 8;
  uint4 lo = *(const uint4*)p;          // K = base .. base+7
  uint4 hi = *(const uint4*)(p + 16);   // K = base+16 .. base+23
  v8u v = {lo.x, lo.y, lo.z, lo.w, hi.x, hi.y, hi.z, hi.w};
  return __builtin_bit_cast(v16bf, v);
}

// B fragment (32x16 bf16) from a K-transposed source X[n][k] (ld elements):
// lane<16 holds col N=lane with K in {0..7,16..23}; lane>=16 K in {8..15,24..31}.
static __device__ __forceinline__ v16bf load_bT_frag(
    const unsigned short* X, int ld, int col0, int k0, int lane) {
  const unsigned short* p =
      X + (size_t)(col0 + (lane & 15)) * ld + k0 + ((lane >> 4) & 1) * 8;
  uint4 lo = *(const uint4*)p;
  uint4 hi = *(const uint4*)(p + 16);
  v8u v = {lo.x, lo.y, lo.z, lo.w, hi.x, hi.y, hi.z, hi.w};
  return __builtin_bit_cast(v16bf, v);
}

// ---------------------------------------------------------------------------
// Tiled bf16 GEMM: Out[M][N] = A[M][K] * B[K][N] + bias[N]
// mode 0: f32 out; mode 1: bf16 out; mode 2: exact GELU then bf16.
// Block 256 thr (8 waves), tile 128x128, K-slab 32, double-buffered LDS.
// A staged with GLOBAL_LOAD_ASYNC_TO_LDS_B128; B staged transposed via regs.
// ---------------------------------------------------------------------------
__global__ __launch_bounds__(256) void gemm_bf16_kernel(
    const unsigned short* __restrict__ A, const unsigned short* __restrict__ B,
    const float* __restrict__ bias, void* __restrict__ Out,
    int M, int N, int K, int mode) {
  __shared__ __align__(16) unsigned short As[2][128 * 32];   // [m][k]
  __shared__ __align__(16) unsigned short BsT[2][128 * 32];  // [n][k]
  const int tid  = threadIdx.x;
  const int lane = tid & 31;
  const int wave = tid >> 5;
  const int wm = wave >> 2, wn = wave & 3;
  const int bm = blockIdx.y * 128, bn = blockIdx.x * 128;
  const int lane15 = lane & 15;
  const int halfsel = (lane >> 4) & 1;

  // staging maps: A 2 thr/row x 16 elems; B 2 k-rows x 8 cols per thread
  const int ar = tid >> 1, ac = (tid & 1) * 16;
  const int bk = (tid >> 4) * 2, bc = (tid & 15) * 8;
  const unsigned long long abase = (unsigned long long)(uintptr_t)A;
  const unsigned avoff0 = ((unsigned)(bm + ar) * (unsigned)K + (unsigned)ac) * 2u;

  v8f acc[4][2];
#pragma unroll
  for (int i = 0; i < 4; ++i)
#pragma unroll
    for (int j = 0; j < 2; ++j)
#pragma unroll
      for (int r = 0; r < 8; ++r) acc[i][j][r] = 0.f;

  auto stage = [&](int s, int p) {
    const int k0 = s << 5;
    // A: two async 16B copies straight into LDS
    const unsigned lds_a = (unsigned)(uintptr_t)(&As[p][ar * 32 + ac]);
    const unsigned vo = avoff0 + (unsigned)(k0 * 2);
    async_ld_b128(lds_a, abase, vo);
    async_ld_b128(lds_a + 16u, abase, vo + 16u);
    // B: load 2x8 block, pack k-pairs, store transposed [n][k]
    const unsigned short* src = B + (size_t)(k0 + bk) * N + bn + bc;
    uint4 r0 = *(const uint4*)src;
    uint4 r1 = *(const uint4*)(src + N);
    const unsigned e0[4] = {r0.x, r0.y, r0.z, r0.w};
    const unsigned e1[4] = {r1.x, r1.y, r1.z, r1.w};
#pragma unroll
    for (int j = 0; j < 8; ++j) {
      const unsigned lo = (j & 1) ? (e0[j >> 1] >> 16) : (e0[j >> 1] & 0xffffu);
      const unsigned hi = (j & 1) ? (e1[j >> 1] & 0xffff0000u) : (e1[j >> 1] << 16);
      *(unsigned*)(&BsT[p][(bc + j) * 32 + bk]) = lo | hi;
    }
  };

  const int nslab = K >> 5;
  stage(0, 0);
  wait_async0();
  __syncthreads();

  for (int s = 0; s < nslab; ++s) {
    const int p = s & 1;
    if (s + 1 < nslab) stage(s + 1, p ^ 1);   // overlap with WMMA below
    const unsigned short* Ab = &As[p][0];
    const unsigned short* Bb = &BsT[p][0];
    v16bf bfrag[2];
#pragma unroll
    for (int nj = 0; nj < 2; ++nj)
      bfrag[nj] = load_bT_frag(Bb, 32, wn * 32 + nj * 16, 0, lane);
#pragma unroll
    for (int mi = 0; mi < 4; ++mi) {
      v16bf afrag = load_a_frag(Ab, 32, wm * 64 + mi * 16, 0, lane);
#pragma unroll
      for (int nj = 0; nj < 2; ++nj)
        acc[mi][nj] = wmma_bf16(afrag, bfrag[nj], acc[mi][nj]);
    }
    if (s + 1 < nslab) wait_async0();
    __syncthreads();
  }

#pragma unroll
  for (int nj = 0; nj < 2; ++nj) {
    const int col = bn + wn * 32 + nj * 16 + lane15;
    const float bc2 = bias[col];
#pragma unroll
    for (int mi = 0; mi < 4; ++mi) {
      const int row0 = bm + wm * 64 + mi * 16 + halfsel * 8;
#pragma unroll
      for (int r = 0; r < 8; ++r) {
        float v = acc[mi][nj][r] + bc2;
        if (mode == 2) v = 0.5f * v * (1.f + erff(v * 0.70710678118654752f));
        const size_t idx = (size_t)(row0 + r) * N + col;
        if (mode == 0) ((float*)Out)[idx] = v;
        else           ((unsigned short*)Out)[idx] = f2bf_raw(v);
      }
    }
  }
}

// ---------------------------------------------------------------------------
// Axial attention: one wave per (b, line, head). L=64 tokens, d=32.
// S = q*k^T + mask (mask as WMMA C operand), row softmax via xor-shuffle
// (xor<16 stays inside the 16-lane half owning a row), P->LDS bf16,
// V staged transposed in LDS (vectorized global loads), O = P*V via two
// K=32 WMMA steps, O written back through LDS with b128 global stores.
// ---------------------------------------------------------------------------
__global__ __launch_bounds__(128) void attn_axis_kernel(
    const unsigned short* __restrict__ qkvbuf,   // [B*H*W][768] q|k|v bf16
    const unsigned short* __restrict__ vsrc,     // axis1: vo_w [B*H*W][256]
    const float* __restrict__ mask,              // [8][64][64]
    unsigned short* __restrict__ out,            // [B*H*W][256] bf16
    int axis) {
  __shared__ __align__(16) unsigned short Pl[4][64 * 64];  // probs per wave
  __shared__ __align__(16) unsigned short Vt[4][32 * 64];  // V^T, reused for O
  const int lane = threadIdx.x & 31;
  const int wave = threadIdx.x >> 5;
  const int lane15 = lane & 15;
  const int halfsel = (lane >> 4) & 1;
  const int head = blockIdx.x * 4 + wave;
  const int line = blockIdx.y;
  const int b = blockIdx.z;

  size_t pix0;
  int pixstride, vstride;
  const unsigned short* vp;
  if (axis == 0) {
    pix0 = ((size_t)b * 64 + line) * 64;  pixstride = 1;
    vp = qkvbuf + pix0 * 768 + 512 + head * 32;  vstride = 768;
  } else {
    pix0 = (size_t)b * 4096 + line;       pixstride = 64;
    vp = vsrc + pix0 * 256 + head * 32;          vstride = 64 * 256;
  }
  const unsigned short* qp = qkvbuf + pix0 * 768 + head * 32;
  const unsigned short* kp = qp + 256;
  const int qstride = pixstride * 768;
  const float* mh = mask + head * 4096;

  // ---- stage V transposed into LDS: Vt[d][pix], lane owns pixel rows 2L,2L+1
  unsigned short* vt = &Vt[wave][0];
  {
    const unsigned short* r0p = vp + (size_t)(2 * lane) * vstride;
    const unsigned short* r1p = r0p + vstride;
    uint4 a0 = *(const uint4*)(r0p);      uint4 a1 = *(const uint4*)(r0p + 8);
    uint4 a2 = *(const uint4*)(r0p + 16); uint4 a3 = *(const uint4*)(r0p + 24);
    uint4 b0 = *(const uint4*)(r1p);      uint4 b1 = *(const uint4*)(r1p + 8);
    uint4 b2 = *(const uint4*)(r1p + 16); uint4 b3 = *(const uint4*)(r1p + 24);
    const unsigned ea[16] = {a0.x, a0.y, a0.z, a0.w, a1.x, a1.y, a1.z, a1.w,
                             a2.x, a2.y, a2.z, a2.w, a3.x, a3.y, a3.z, a3.w};
    const unsigned eb[16] = {b0.x, b0.y, b0.z, b0.w, b1.x, b1.y, b1.z, b1.w,
                             b2.x, b2.y, b2.z, b2.w, b3.x, b3.y, b3.z, b3.w};
    unsigned* vtw = (unsigned*)vt;
#pragma unroll
    for (int d = 0; d < 32; ++d) {
      const unsigned lo = (d & 1) ? (ea[d >> 1] >> 16) : (ea[d >> 1] & 0xffffu);
      const unsigned hi = (d & 1) ? (eb[d >> 1] & 0xffff0000u) : (eb[d >> 1] << 16);
      vtw[d * 32 + lane] = lo | hi;   // covers pixels 2L,2L+1 of row d
    }
  }

  // ---- scores + mask
  v8f S[4][4];
#pragma unroll
  for (int mi = 0; mi < 4; ++mi) {
    v16bf a = load_a_frag(qp, qstride, mi * 16, 0, lane);
#pragma unroll
    for (int nj = 0; nj < 4; ++nj) {
      v8f c;
      const int rbase = mi * 16 + halfsel * 8;
      const int col = nj * 16 + lane15;
#pragma unroll
      for (int r = 0; r < 8; ++r) c[r] = mh[(rbase + r) * 64 + col];
      v16bf bT = load_bT_frag(kp, qstride, nj * 16, 0, lane);
      S[mi][nj] = wmma_bf16(a, bT, c);
    }
  }
  // ---- softmax along each row (fast hardware exp)
#pragma unroll
  for (int mi = 0; mi < 4; ++mi) {
#pragma unroll
    for (int r = 0; r < 8; ++r) {
      float m = S[mi][0][r];
#pragma unroll
      for (int nj = 1; nj < 4; ++nj) m = fmaxf(m, S[mi][nj][r]);
#pragma unroll
      for (int o = 1; o < 16; o <<= 1) m = fmaxf(m, __shfl_xor(m, o, 32));
      float s = 0.f;
#pragma unroll
      for (int nj = 0; nj < 4; ++nj) {
        float e = __expf(S[mi][nj][r] - m);
        S[mi][nj][r] = e;
        s += e;
      }
#pragma unroll
      for (int o = 1; o < 16; o <<= 1) s += __shfl_xor(s, o, 32);
      const float inv = 1.f / s;
#pragma unroll
      for (int nj = 0; nj < 4; ++nj) S[mi][nj][r] *= inv;
    }
  }
  // ---- P -> LDS (bf16), private per wave
  unsigned short* pl = &Pl[wave][0];
#pragma unroll
  for (int mi = 0; mi < 4; ++mi)
#pragma unroll
    for (int nj = 0; nj < 4; ++nj)
#pragma unroll
      for (int r = 0; r < 8; ++r)
        pl[(mi * 16 + halfsel * 8 + r) * 64 + nj * 16 + lane15] =
            f2bf_raw(S[mi][nj][r]);
  // ---- O = P * V  (M=64,N=32,K=64); V frags from Vt, contiguous b128 pairs
  v8f O[4][2];
#pragma unroll
  for (int mi = 0; mi < 4; ++mi)
#pragma unroll
    for (int nj = 0; nj < 2; ++nj)
#pragma unroll
      for (int r = 0; r < 8; ++r) O[mi][nj][r] = 0.f;
#pragma unroll
  for (int ks = 0; ks < 2; ++ks) {
    v16bf bfrag[2];
#pragma unroll
    for (int nj = 0; nj < 2; ++nj)
      bfrag[nj] = load_bT_frag(vt, 64, nj * 16, ks * 32, lane);
#pragma unroll
    for (int mi = 0; mi < 4; ++mi) {
      v16bf a = load_a_frag(pl, 64, mi * 16, ks * 32, lane);
#pragma unroll
      for (int nj = 0; nj < 2; ++nj)
        O[mi][nj] = wmma_bf16(a, bfrag[nj], O[mi][nj]);
    }
  }
  // ---- O -> LDS tile (reuse Vt: all Vt reads are program-order earlier and
  // LDS is in-order per wave), then vectorized b128 writeback
  unsigned short* ot = vt;   // [row=64][d=32]
#pragma unroll
  for (int mi = 0; mi < 4; ++mi)
#pragma unroll
    for (int nj = 0; nj < 2; ++nj)
#pragma unroll
      for (int r = 0; r < 8; ++r)
        ot[(mi * 16 + halfsel * 8 + r) * 32 + nj * 16 + lane15] =
            f2bf_raw(O[mi][nj][r]);
#pragma unroll
  for (int rr = 0; rr < 2; ++rr) {
    const int row = 2 * lane + rr;
    const uint4* src = (const uint4*)(ot + row * 32);
    uint4 d0 = src[0], d1 = src[1];
    unsigned short* dst =
        out + (pix0 + (size_t)row * pixstride) * 256 + head * 32;
    *(uint4*)dst = d0;
    *(uint4*)(dst + 8) = d1;
  }
}

// ---------------------------------------------------------------------------
// LayerNorm over C=256 -> bf16; one wave per pixel, 8 waves per block.
// ---------------------------------------------------------------------------
__global__ __launch_bounds__(256) void ln_bf16_kernel(
    const float* __restrict__ x, const float* __restrict__ g,
    const float* __restrict__ bt, unsigned short* __restrict__ out) {
  const int lane = threadIdx.x & 31, wave = threadIdx.x >> 5;
  const size_t pix = (size_t)blockIdx.x * 8 + wave;
  const float* xp = x + pix * 256 + lane * 8;
  float4 v0 = *(const float4*)xp;
  float4 v1 = *(const float4*)(xp + 4);
  float vals[8] = {v0.x, v0.y, v0.z, v0.w, v1.x, v1.y, v1.z, v1.w};
  float s = 0.f, ss = 0.f;
#pragma unroll
  for (int i = 0; i < 8; ++i) { s += vals[i]; ss += vals[i] * vals[i]; }
#pragma unroll
  for (int o = 1; o < 32; o <<= 1) {
    s  += __shfl_xor(s, o, 32);
    ss += __shfl_xor(ss, o, 32);
  }
  const float mean = s * (1.f / 256.f);
  const float var  = ss * (1.f / 256.f) - mean * mean;
  const float inv  = rsqrtf(var + 1e-6f);
#pragma unroll
  for (int i = 0; i < 8; ++i) {
    const int c = lane * 8 + i;
    out[pix * 256 + c] = f2bf_raw((vals[i] - mean) * inv * g[c] + bt[c]);
  }
}

// ---------------------------------------------------------------------------
// 3x3 depthwise conv (SAME) + bias + residual (f32).
// ---------------------------------------------------------------------------
__global__ __launch_bounds__(256) void dwconv3_res_kernel(
    const float* __restrict__ x, const float* __restrict__ w,
    const float* __restrict__ bias, float* __restrict__ out) {
  const size_t i = (size_t)blockIdx.x * 256 + threadIdx.x;
  const int c  = (int)(i & 255);
  const size_t p = i >> 8;
  const int xw = (int)(p & 63);
  const int yh = (int)((p >> 6) & 63);
  const int bb = (int)(p >> 12);
  float acc = bias[c];
#pragma unroll
  for (int ky = 0; ky < 3; ++ky) {
    const int yy = yh + ky - 1;
    if (yy < 0 || yy > 63) continue;
#pragma unroll
    for (int kx = 0; kx < 3; ++kx) {
      const int xx = xw + kx - 1;
      if (xx < 0 || xx > 63) continue;
      acc += x[(((size_t)bb * 64 + yy) * 64 + xx) * 256 + c] *
             w[(ky * 3 + kx) * 256 + c];
    }
  }
  out[i] = x[i] + acc;
}

// ---------------------------------------------------------------------------
// attn_in = bf16( vo_h + dwconv5x5(v) + lepe_b ), v from qkv buffer (bf16).
// ---------------------------------------------------------------------------
__global__ __launch_bounds__(256) void lepe_add_kernel(
    const unsigned short* __restrict__ voh,
    const unsigned short* __restrict__ qkvbuf,
    const float* __restrict__ w, const float* __restrict__ bias,
    unsigned short* __restrict__ out) {
  const size_t i = (size_t)blockIdx.x * 256 + threadIdx.x;
  const int c  = (int)(i & 255);
  const size_t p = i >> 8;
  const int xw = (int)(p & 63);
  const int yh = (int)((p >> 6) & 63);
  const int bb = (int)(p >> 12);
  float acc = bias[c];
#pragma unroll
  for (int ky = 0; ky < 5; ++ky) {
    const int yy = yh + ky - 2;
    if (yy < 0 || yy > 63) continue;
#pragma unroll
    for (int kx = 0; kx < 5; ++kx) {
      const int xx = xw + kx - 2;
      if (xx < 0 || xx > 63) continue;
      acc += bf2f(qkvbuf[(((size_t)bb * 64 + yy) * 64 + xx) * 768 + 512 + c]) *
             w[(ky * 5 + kx) * 256 + c];
    }
  }
  out[i] = f2bf_raw(bf2f(voh[i]) + acc);
}

// ---------------------------------------------------------------------------
// Elementwise helpers
// ---------------------------------------------------------------------------
__global__ __launch_bounds__(256) void add_inplace_kernel(
    float* __restrict__ a, const float* __restrict__ b2, int n) {
  const int i = blockIdx.x * 256 + threadIdx.x;
  if (i < n) a[i] += b2[i];
}
__global__ __launch_bounds__(256) void add_out_kernel(
    const float* __restrict__ a, const float* __restrict__ b2,
    float* __restrict__ o, int n) {
  const int i = blockIdx.x * 256 + threadIdx.x;
  if (i < n) o[i] = a[i] + b2[i];
}
__global__ __launch_bounds__(256) void cast_bf_kernel(
    const float* __restrict__ in, unsigned short* __restrict__ out, int n) {
  const int i = blockIdx.x * 256 + threadIdx.x;
  if (i < n) out[i] = f2bf_raw(in[i]);
}
// Pack wq | wk*scale | wv into one bf16 [256][768] and fold scale into bk.
__global__ __launch_bounds__(256) void pack_qkvw_kernel(
    const float* __restrict__ wq, const float* __restrict__ wk,
    const float* __restrict__ wv, const float* __restrict__ bq,
    const float* __restrict__ bk, const float* __restrict__ bv,
    unsigned short* __restrict__ wout, float* __restrict__ bout, float scale) {
  const int i = blockIdx.x * 256 + threadIdx.x;  // 256*768
  const int k = i / 768, j = i % 768;
  float v;
  if (j < 256)      v = wq[k * 256 + j];
  else if (j < 512) v = wk[k * 256 + j - 256] * scale;
  else              v = wv[k * 256 + j - 512];
  wout[i] = f2bf_raw(v);
  if (k == 0) {
    bout[j] = (j < 256) ? bq[j] : (j < 512) ? bk[j - 256] * scale : bv[j - 512];
  }
}

// ---------------------------------------------------------------------------
// Host orchestration
// ---------------------------------------------------------------------------
extern "C" void kernel_launch(void* const* d_in, const int* in_sizes, int n_in,
                              void* d_out, int out_size, void* d_ws,
                              size_t ws_size, hipStream_t stream) {
  const float* x      = (const float*)d_in[0];
  const float* mask_h = (const float*)d_in[1];
  const float* mask_w = (const float*)d_in[2];
  const float* pos_w  = (const float*)d_in[3];
  const float* pos_b  = (const float*)d_in[4];
  const float* ln1_g  = (const float*)d_in[5];
  const float* ln1_b  = (const float*)d_in[6];
  const float* wq = (const float*)d_in[7];  const float* bq = (const float*)d_in[8];
  const float* wk = (const float*)d_in[9];  const float* bk = (const float*)d_in[10];
  const float* wv = (const float*)d_in[11]; const float* bv = (const float*)d_in[12];
  const float* lepe_w = (const float*)d_in[13]; const float* lepe_b = (const float*)d_in[14];
  const float* wo = (const float*)d_in[15]; const float* bo = (const float*)d_in[16];
  const float* ln2_g = (const float*)d_in[17]; const float* ln2_b = (const float*)d_in[18];
  const float* fc1_w = (const float*)d_in[19]; const float* fc1_b = (const float*)d_in[20];
  const float* fc2_w = (const float*)d_in[21]; const float* fc2_b = (const float*)d_in[22];

  // B=16, H=W=64, C=256, F=1024; M = 65536 pixels. All ws regions 256B-aligned.
  char* W = (char*)d_ws;
  float*          x1      = (float*)(W + 0);                   //  67,108,864
  unsigned short* lnbuf   = (unsigned short*)(W + 67108864);   //  33,554,432
  unsigned short* qkv     = (unsigned short*)(W + 100663296);  // 100,663,296
  unsigned short* vo_w    = (unsigned short*)(W + 201326592);  //  33,554,432
  unsigned short* h1      = qkv;                               // alias (qkv+vo_w dead)
  unsigned short* vo_h    = (unsigned short*)(W + 234881024);  //  33,554,432
  unsigned short* attnin  = (unsigned short*)(W + 268435456);  //  33,554,432
  float*          gout    = (float*)(W + 301989888);           //  67,108,864
  unsigned short* wqkv_bf = (unsigned short*)(W + 369098752);  //     393,216
  float*          bqkv    = (float*)(W + 369491968);           //       3,072
  unsigned short* wo_bf   = (unsigned short*)(W + 369495296);  //     131,072
  unsigned short* fc1_bf  = (unsigned short*)(W + 369626368);  //     524,288
  unsigned short* fc2_bf  = (unsigned short*)(W + 370150656);  //     524,288
  (void)in_sizes; (void)n_in; (void)out_size; (void)ws_size;

  const float kscale = 0.1767766952966369f;  // 32^-0.5

  // weight prep
  pack_qkvw_kernel<<<768, 256, 0, stream>>>(wq, wk, wv, bq, bk, bv,
                                            wqkv_bf, bqkv, kscale);
  cast_bf_kernel<<<256, 256, 0, stream>>>(wo, wo_bf, 65536);
  cast_bf_kernel<<<1024, 256, 0, stream>>>(fc1_w, fc1_bf, 262144);
  cast_bf_kernel<<<1024, 256, 0, stream>>>(fc2_w, fc2_bf, 262144);

  // x1 = x + dwconv3(x)
  dwconv3_res_kernel<<<65536, 256, 0, stream>>>(x, pos_w, pos_b, x1);
  // ln1 -> bf16
  ln_bf16_kernel<<<8192, 256, 0, stream>>>(x1, ln1_g, ln1_b, lnbuf);
  // fused qkv projection: [65536 x 768] = ln1 @ [256 x 768]
  gemm_bf16_kernel<<<dim3(6, 512), 256, 0, stream>>>(
      lnbuf, wqkv_bf, bqkv, (void*)qkv, 65536, 768, 256, 1);
  // W-axis then H-axis attention
  attn_axis_kernel<<<dim3(2, 64, 16), 128, 0, stream>>>(qkv, vo_w, mask_w, vo_w, 0);
  attn_axis_kernel<<<dim3(2, 64, 16), 128, 0, stream>>>(qkv, vo_w, mask_h, vo_h, 1);
  // attn_in = vo_h + lepe(v)
  lepe_add_kernel<<<65536, 256, 0, stream>>>(vo_h, qkv, lepe_w, lepe_b, attnin);
  // output projection (f32 out)
  gemm_bf16_kernel<<<dim3(2, 512), 256, 0, stream>>>(
      attnin, wo_bf, bo, (void*)gout, 65536, 256, 256, 0);
  // x1 += proj
  add_inplace_kernel<<<65536, 256, 0, stream>>>(x1, gout, 16777216);
  // ln2 -> bf16
  ln_bf16_kernel<<<8192, 256, 0, stream>>>(x1, ln2_g, ln2_b, lnbuf);
  // fc1 + exact GELU -> bf16 (h1 aliases dead qkv region)
  gemm_bf16_kernel<<<dim3(8, 512), 256, 0, stream>>>(
      lnbuf, fc1_bf, fc1_b, (void*)h1, 65536, 1024, 256, 2);
  // fc2 -> f32
  gemm_bf16_kernel<<<dim3(2, 512), 256, 0, stream>>>(
      h1, fc2_bf, fc2_b, (void*)gout, 65536, 256, 1024, 0);
  // out = x1 + fc2
  add_out_kernel<<<65536, 256, 0, stream>>>(x1, gout, (float*)d_out, 16777216);
}
